// MNIST_net_24885040513056
// MI455X (gfx1250) — matrix operand order repaired and checked by reading, not compiled, not run on hardware
//
#include <hip/hip_runtime.h>

typedef __attribute__((ext_vector_type(16))) _Float16 v16h;
typedef __attribute__((ext_vector_type(8)))  float    v8f;
typedef __attribute__((ext_vector_type(4)))  float    v4f;
typedef __attribute__((ext_vector_type(2)))  int      v2i;

#define BATCH 8
#define KC 10
#define KADJ 4
#define N0 327680
#define N1 81920
#define N2 20480

// ds_swizzle_b32 group-of-32 XOR patterns: offset = xor<<10 | and(0x1f)
#define SWZ_XOR1 0x041f
#define SWZ_XOR2 0x081f

__device__ __forceinline__ float lane_max4(float v) {
    float t1 = __int_as_float(__builtin_amdgcn_ds_swizzle(__float_as_int(v), SWZ_XOR1));
    v = fmaxf(v, t1);
    float t2 = __int_as_float(__builtin_amdgcn_ds_swizzle(__float_as_int(v), SWZ_XOR2));
    return fmaxf(v, t2);
}

// Contraction permutation (PERM layers): B element (ch,hi,j) covers
//   q = ch*16 + j,  kconv = q % KC  (constant per element),  c = 2*(q/KC) + hi.
// Bijective onto ck = c*KC + kconv when CK % 32 == 0 and CIN even.
template<int CK, int CIN>
__device__ __forceinline__ constexpr bool use_perm() {
    return (CK % 32) == 0 && (CIN % 2) == 0;
}

// ---------------------------------------------------------------------------
// Pre-pack weights into WMMA A-matrix register layout, f16, zero-padded,
// applying the same contraction permutation as the B gather.
// Flat half index: (((mtile*NCHUNK + ch)*2 + hi)*16 + col)*16 + j,
// element j of lane (hi,col) covers logical k = ch*32 + hi*16 + j.
// ---------------------------------------------------------------------------
template<int CK, int MOUT, int MTILES, int CIN>
__global__ __launch_bounds__(256)
void pack_weights_kernel(const float* __restrict__ w, _Float16* __restrict__ wp)
{
    constexpr int NCHUNK = (CK + 31) / 32;
    constexpr int TOTAL  = MTILES * NCHUNK * 512;
    constexpr bool PERM  = use_perm<CK, CIN>();
    const int idx = blockIdx.x * 256 + threadIdx.x;
    if (idx >= TOTAL) return;
    const int j     = idx & 15;
    const int col   = (idx >> 4) & 15;
    const int hi    = (idx >> 8) & 1;
    const int mc    = idx >> 9;
    const int ch    = mc % NCHUNK;
    const int mtile = mc / NCHUNK;
    const int k = ch * 32 + hi * 16 + j;   // logical K position
    int ck;
    if (PERM) {
        const int kk = k & 31;
        const int hib = kk >> 4;
        const int jb  = kk & 15;
        const int q   = (k >> 5) * 16 + jb;
        ck = (2 * (q / KC) + hib) * KC + (q % KC);
    } else {
        ck = k;
    }
    const int o = mtile * 16 + col;
    float val = 0.0f;
    if (ck < CK && o < MOUT) val = w[(size_t)o * CK + ck];
    wp[idx] = (_Float16)val;
}

// ---------------------------------------------------------------------------
// Shared B-tile gather: 16 f16 elements for this lane's column.
// ---------------------------------------------------------------------------
template<int CK, int CIN>
__device__ __forceinline__ v16h gather_b(const float* __restrict__ xb,
                                         const int* __restrict__ tidx,
                                         int ch, int hi, int hiN, int Nin)
{
    constexpr bool PERM = use_perm<CK, CIN>();
    v16h bmat;
    #pragma unroll
    for (int j = 0; j < 16; ++j) {
        if (PERM) {
            const int q    = ch * 16 + j;
            const int kidx = q % KC;              // compile-time constant
            const int r    = q / KC;              // compile-time constant
            const int off  = 2 * r * Nin;         // wave-uniform (SALU)
            bmat[j] = (_Float16)xb[off + hiN + tidx[kidx]];
        } else {
            const int k0 = ch * 32 + j;
            const int k1 = ch * 32 + 16 + j;
            const int kidx0 = k0 % KC;
            const int kidx1 = k1 % KC;
            const int cidx0 = (k0 / KC < CIN) ? k0 / KC : CIN - 1;
            const int cidx1 = (k1 / KC < CIN) ? k1 / KC : CIN - 1;
            const int tt   = hi ? tidx[kidx1] : tidx[kidx0];
            const int xoff = hi ? cidx1 * Nin : cidx0 * Nin;
            float val = xb[xoff + tt];
            if ((hi ? k1 : k0) >= CK) val = 0.0f;
            bmat[j] = (_Float16)val;
        }
    }
    return bmat;
}

__device__ __forceinline__ void load_tidx(const int* __restrict__ table,
                                          int node, int* __restrict__ tidx)
{
    const v2i* tp = (const v2i*)(table + (size_t)node * KC);
    #pragma unroll
    for (int u = 0; u < KC / 2; ++u) {
        const v2i pr = tp[u];
        tidx[2 * u]     = pr[0];
        tidx[2 * u + 1] = pr[1];
    }
}

// ---------------------------------------------------------------------------
// Fused graph-conv + maxpool:
//   out[b,o,i] = max_{j<4} sum_ck w[o,ck] * x[b, ck/KC, table[adj[pool[i],j], ck%KC]]
// One wave32 per (mtile, T tiles of 4 pooled outputs); b uniform per block.
// ---------------------------------------------------------------------------
template<int CK, int MOUT, int CIN, int NTILES, int MTILES, int T>
__global__ __launch_bounds__(256)
void conv_pool_wmma_kernel(const float*    __restrict__ x,
                           const int*      __restrict__ table,
                           const int*      __restrict__ pool,
                           const int*      __restrict__ adj,
                           const _Float16* __restrict__ wpack,
                           const float*    __restrict__ bias,
                           float*          __restrict__ out,
                           int Nin, int Nout)
{
    static_assert(MOUT % 16 == 0, "fused kernel requires full M tiles");
    static_assert((MTILES * NTILES) % 8 == 0, "grid must divide");
    constexpr int NCHUNK = (CK + 31) / 32;
    constexpr int BLOCKS_PER_B = (MTILES * NTILES) / 8;

    const int lane = threadIdx.x & 31;
    const int wid  = threadIdx.x >> 5;
    const int hi   = lane >> 4;
    const int col  = lane & 15;

    const int b        = blockIdx.x / BLOCKS_PER_B;          // SALU-uniform
    const int tile_lin = (blockIdx.x % BLOCKS_PER_B) * 8 + wid;
    const int mtile    = tile_lin / NTILES;
    const int ntile    = tile_lin % NTILES;

    const float* xb  = x + (size_t)b * CIN * Nin;            // uniform base (saddr)
    const v16h*  wv  = (const v16h*)wpack;
    const int    hiN = hi * Nin;

    int i[T], node[T], tidx[T][KC];
    #pragma unroll
    for (int t = 0; t < T; ++t) {
        i[t] = (ntile * T + t) * 4 + (col >> 2);
        const int pidx = pool[i[t]];
        node[t] = adj[(size_t)pidx * KADJ + (col & 3)];
        load_tidx(table, node[t], tidx[t]);
    }

    v8f acc[T];
    #pragma unroll
    for (int t = 0; t < T; ++t) acc[t] = (v8f){};

    #pragma unroll
    for (int ch = 0; ch < NCHUNK; ++ch) {
        const v16h amat = wv[((mtile * NCHUNK + ch) * 2 + hi) * 16 + col];
        #pragma unroll
        for (int t = 0; t < T; ++t) {
            const v16h bmat = gather_b<CK, CIN>(xb, tidx[t], ch, hi, hiN, Nin);
            acc[t] = __builtin_amdgcn_wmma_f32_16x16x32_f16(
                        false, amat, false, bmat, (short)0, acc[t], false, false);
        }
    }

    const v4f* bv = (const v4f*)(bias + mtile * 16 + hi * 8);
    const v4f blo = bv[0], bhi = bv[1];

    #pragma unroll
    for (int t = 0; t < T; ++t) {
        float res[8];
        #pragma unroll
        for (int vi = 0; vi < 8; ++vi) {
            const float bb = (vi < 4) ? blo[vi & 3] : bhi[vi & 3];
            res[vi] = lane_max4(acc[t][vi] + bb);
        }
        if ((lane & 3) == 0) {
            const size_t base =
                ((size_t)b * MOUT + mtile * 16 + hi * 8) * (size_t)Nout + i[t];
            #pragma unroll
            for (int vi = 0; vi < 8; ++vi)
                out[base + (size_t)vi * Nout] = res[vi];
        }
    }
}

// ---------------------------------------------------------------------------
// Plain graph-conv (layer 3, M padded to 16 in wpack): one wave per T tiles.
// ---------------------------------------------------------------------------
template<int CK, int MOUT, int CIN, int NTILES, int T>
__global__ __launch_bounds__(256)
void conv_wmma_kernel(const float*    __restrict__ x,
                      const int*      __restrict__ table,
                      const _Float16* __restrict__ wpack,
                      const float*    __restrict__ bias,
                      float*          __restrict__ out,
                      int N)
{
    static_assert(NTILES % 8 == 0, "grid must divide");
    constexpr int NCHUNK = (CK + 31) / 32;
    constexpr int BLOCKS_PER_B = NTILES / 8;

    const int lane = threadIdx.x & 31;
    const int wid  = threadIdx.x >> 5;
    const int hi   = lane >> 4;
    const int col  = lane & 15;

    const int b     = blockIdx.x / BLOCKS_PER_B;             // SALU-uniform
    const int ntile = (blockIdx.x % BLOCKS_PER_B) * 8 + wid;

    const float* xb  = x + (size_t)b * CIN * N;
    const v16h*  wv  = (const v16h*)wpack;
    const int    hiN = hi * N;

    int n[T], tidx[T][KC];
    #pragma unroll
    for (int t = 0; t < T; ++t) {
        n[t] = (ntile * T + t) * 16 + col;
        load_tidx(table, n[t], tidx[t]);
    }

    v8f acc[T];
    #pragma unroll
    for (int t = 0; t < T; ++t) acc[t] = (v8f){};

    #pragma unroll
    for (int ch = 0; ch < NCHUNK; ++ch) {
        const v16h amat = wv[(ch * 2 + hi) * 16 + col];
        #pragma unroll
        for (int t = 0; t < T; ++t) {
            const v16h bmat = gather_b<CK, CIN>(xb, tidx[t], ch, hi, hiN, N);
            acc[t] = __builtin_amdgcn_wmma_f32_16x16x32_f16(
                        false, amat, false, bmat, (short)0, acc[t], false, false);
        }
    }

    #pragma unroll
    for (int t = 0; t < T; ++t) {
        #pragma unroll
        for (int vi = 0; vi < 8; ++vi) {
            const int o = vi + hi * 8;
            if (o < MOUT)
                out[((size_t)b * MOUT + o) * (size_t)N + n[t]] = acc[t][vi] + bias[o];
        }
    }
}

// ---------------------------------------------------------------------------
// out[bo] = mean_n in[bo, n]   (one block per (b,o))
// ---------------------------------------------------------------------------
__global__ __launch_bounds__(256)
void mean_kernel(const float* __restrict__ in, float* __restrict__ out, int N)
{
    __shared__ float red[256];
    const int bo = blockIdx.x;
    float s = 0.0f;
    for (int idx = threadIdx.x; idx < N; idx += 256)
        s += in[(size_t)bo * N + idx];
    red[threadIdx.x] = s;
    __syncthreads();
    for (int st = 128; st > 0; st >>= 1) {
        if (threadIdx.x < st) red[threadIdx.x] += red[threadIdx.x + st];
        __syncthreads();
    }
    if (threadIdx.x == 0) out[bo] = red[0] / (float)N;
}

extern "C" void kernel_launch(void* const* d_in, const int* in_sizes, int n_in,
                              void* d_out, int out_size, void* d_ws, size_t ws_size,
                              hipStream_t stream) {
    const float* x   = (const float*)d_in[0];
    const int* t0    = (const int*)d_in[1];
    const int* t1    = (const int*)d_in[2];
    const int* t2    = (const int*)d_in[3];
    const int* adj0  = (const int*)d_in[4];
    const int* adj1  = (const int*)d_in[5];
    const int* pool0 = (const int*)d_in[6];
    const int* pool1 = (const int*)d_in[7];
    const float* w1  = (const float*)d_in[8];
    const float* b1  = (const float*)d_in[9];
    const float* w2  = (const float*)d_in[10];
    const float* b2  = (const float*)d_in[11];
    const float* w3  = (const float*)d_in[12];
    const float* b3  = (const float*)d_in[13];
    float* outp = (float*)d_out;

    // Workspace: p1 | p2 | h3 | packed weights  (~70 MB, fully L2-resident)
    char* ws = (char*)d_ws;
    const size_t P1_BYTES = (size_t)BATCH * 16 * N1 * sizeof(float); // 41,943,040
    const size_t P2_BYTES = (size_t)BATCH * 32 * N2 * sizeof(float); // 20,971,520
    const size_t H3_BYTES = (size_t)BATCH * 10 * N2 * sizeof(float); //  6,553,600
    float* p1 = (float*)ws;
    float* p2 = (float*)(ws + P1_BYTES);
    float* h3 = (float*)(ws + P1_BYTES + P2_BYTES);
    char*  wpbase = ws + P1_BYTES + P2_BYTES + H3_BYTES;
    _Float16* wp1 = (_Float16*)(wpbase);                // 512   halves = 1 KB
    _Float16* wp2 = (_Float16*)(wpbase + 1024);         // 5120  halves = 10 KB
    _Float16* wp3 = (_Float16*)(wpbase + 1024 + 10240); // 5120  halves = 10 KB

    // Pre-pack weights into (permuted) WMMA A-layout
    pack_weights_kernel<30, 16, 1, 3><<<2, 256, 0, stream>>>(w1, wp1);
    pack_weights_kernel<160, 32, 2, 16><<<20, 256, 0, stream>>>(w2, wp2);
    pack_weights_kernel<320, 10, 1, 32><<<20, 256, 0, stream>>>(w3, wp3);

    // Fused layer 1 + pool 1: (B,3,N0) -> (B,16,N1). CK=30, 1 M-tile, T=2.
    {
        constexpr int NT = N1 / 8;                  // 10240
        conv_pool_wmma_kernel<30, 16, 3, NT, 1, 2>
            <<<BATCH * (NT / 8), 256, 0, stream>>>(
            x, t0, pool0, adj0, wp1, b1, p1, N0, N1);
    }
    // Fused layer 2 + pool 2: (B,16,N1) -> (B,32,N2). CK=160, 2 M-tiles, T=2.
    {
        constexpr int NT = N2 / 8;                  // 2560
        conv_pool_wmma_kernel<160, 32, 16, NT, 2, 2>
            <<<BATCH * (2 * NT / 8), 256, 0, stream>>>(
            p1, t1, pool1, adj1, wp2, b2, p2, N1, N2);
    }
    // Layer 3: (B,32,N2) -> (B,10,N2). CK=320, M padded to 16, T=2.
    {
        constexpr int NT = N2 / 32;                 // 640
        conv_wmma_kernel<320, 10, 32, NT, 2>
            <<<BATCH * (NT / 8), 256, 0, stream>>>(
            p2, t2, wp3, b3, h3, N2);
    }
    // Mean over nodes: (B,10,N2) -> (B,10)
    mean_kernel<<<BATCH * 10, 256, 0, stream>>>(h3, outp, N2);
}
